// HopfieldCore_74217034875429
// MI455X (gfx1250) — matrix operand order repaired and checked by reading, not compiled, not run on hardware
//
#include <hip/hip_runtime.h>
#include <cstdint>

#define E_  512
#define L_  512
#define S_  4096
#define N_  4
#define H_  8
#define Dh_ 64

typedef __attribute__((ext_vector_type(16))) __bf16        v16bf;
typedef __attribute__((ext_vector_type(8)))  float         v8f;
typedef __attribute__((ext_vector_type(8)))  unsigned int  v8u;
typedef __attribute__((ext_vector_type(8)))  int           v8i;
typedef __attribute__((ext_vector_type(4)))  unsigned int  v4u;
typedef __attribute__((ext_vector_type(4)))  int           v4i;

__device__ __forceinline__ unsigned short f2bf(float f) {
  unsigned u = __float_as_uint(f);
  unsigned r = u + 0x7FFFu + ((u >> 16) & 1u);   // round-to-nearest-even
  return (unsigned short)(r >> 16);
}

__device__ __forceinline__ v8f wmma_bf16f32(v8u a, v8u b, v8f c) {
  return __builtin_amdgcn_wmma_f32_16x16x32_bf16(
      false, __builtin_bit_cast(v16bf, a),
      false, __builtin_bit_cast(v16bf, b),
      (short)0, c, false, false);
}

// low 32 bits of a generic pointer to LDS = byte offset within the workgroup LDS
__device__ __forceinline__ unsigned lds_off(const void* p) {
  return (unsigned)(uintptr_t)p;
}

// TDM: DMA a (d1 rows x d0 8-byte-elems) tile (row stride s0 elems) from global
// into LDS, packed row-major contiguously. D# per CDNA5 ISA.
// 6-arg builtin form (clang-23 / therock-10.0 toolchain).
__device__ __forceinline__ void tdm_load(unsigned lds_addr, const void* gptr,
                                         unsigned d0, unsigned d1, unsigned s0) {
  unsigned long long ga = (unsigned long long)(uintptr_t)gptr;
  v4u g0 = {1u,                                  // count=1, no gather
            lds_addr,                            // lds_addr (bytes)
            (unsigned)ga,                        // global_addr[31:0]
            (unsigned)((ga >> 32) & 0x01FFFFFFu) | (2u << 30)};  // ga[56:32] | type=2
  v8u g1;
  g1[0] = 3u << 16;                              // data_size = 8 bytes
  g1[1] = (s0 & 0xFFFFu) << 16;                  // tensor_dim0[15:0]   (= row length)
  g1[2] = (s0 >> 16) | ((d1 & 0xFFFFu) << 16);   // tensor_dim0[31:16] | tensor_dim1[15:0]
  g1[3] = (d1 >> 16) | (d0 << 16);               // tensor_dim1[31:16] | tile_dim0
  g1[4] = (d1 > 1u) ? d1 : 0u;                   // tile_dim1 (0 = unused) | tile_dim2=0
  g1[5] = s0;                                    // tensor_dim0_stride[31:0]
  g1[6] = 0u;
  g1[7] = 0u;
  v4i z4 = {0, 0, 0, 0};
  v8i z8 = {0, 0, 0, 0, 0, 0, 0, 0};
  __builtin_amdgcn_tensor_load_to_lds(g0, __builtin_bit_cast(v8i, g1), z4, z4, z8, 0);
}

// ---------------- prep: fp32 -> bf16 flat convert ----------------
__global__ __launch_bounds__(256) void cvt_f32_bf16(
    const float* __restrict__ in, unsigned short* __restrict__ out, int n) {
  int i = blockIdx.x * 256 + threadIdx.x;
  if (i < n) out[i] = f2bf(in[i]);
}

// ---------------- bf16 GEMM: C[M x 512] = A[M x 512] @ W^T + bias ----------------
// W passed row-major [n_out][k] bf16 (no transpose needed: LDS tile [n][k] == W tile).
__global__ __launch_bounds__(256) void gemm_bf16(
    const unsigned short* __restrict__ A,
    const unsigned short* __restrict__ W,     // [512 n][512 k] bf16
    const float* __restrict__ bias,
    float* __restrict__ C, int M) {
  const int K = 512, NC = 512;
  __shared__ unsigned short Bst[64 * 32];        // W tile: [n][k] row-major, 4KB
  int tid = threadIdx.x;
  int wave = tid >> 5, lane = tid & 31;
  int half = lane >> 4, l15 = lane & 15;
  int row0 = blockIdx.x * 128 + wave * 16;
  int col0 = blockIdx.y * 64;
  int kb8 = half * 8, kb16 = half * 16;

  v8f zf = {0.f, 0.f, 0.f, 0.f, 0.f, 0.f, 0.f, 0.f};
  v8f acc[4];
  acc[0] = zf; acc[1] = zf; acc[2] = zf; acc[3] = zf;

  const unsigned short* Ar = A + (size_t)(row0 + l15) * K;
  unsigned bst = lds_off(Bst);

  for (int k0 = 0; k0 < K; k0 += 32) {
    __syncthreads();
    if (wave == 0) {
      // 64 rows x 64B (8 x 8B elems), row stride 1024B = 128 elems
      tdm_load(bst, W + (size_t)col0 * K + k0, 8u, 64u, 128u);
      __builtin_amdgcn_s_wait_tensorcnt(0);
    }
    if (k0 + 32 < K) __builtin_prefetch(Ar + k0 + 32, 0, 1);
    __syncthreads();

    // A fragment: lane = row (l15), 16 bf16 of this K chunk per ISA layout
    v8u av;
#pragma unroll
    for (int j = 0; j < 8; ++j) {
      int e = (j < 4) ? (kb8 + 2 * j) : (16 + kb8 + 2 * (j - 4));
      av[j] = *(const unsigned int*)(Ar + k0 + e);
    }
#pragma unroll
    for (int nt = 0; nt < 4; ++nt) {
      v8u bv;
#pragma unroll
      for (int j = 0; j < 8; ++j)
        bv[j] = *(const unsigned int*)&Bst[(nt * 16 + l15) * 32 + ((j < 4) ? (kb8 + 2 * j) : (16 + kb8 + 2 * (j - 4)))];
      acc[nt] = wmma_bf16f32(av, bv, acc[nt]);
    }
  }
#pragma unroll
  for (int nt = 0; nt < 4; ++nt)
#pragma unroll
    for (int r = 0; r < 8; ++r) {
      int rr = r + half * 8;
      int col = col0 + nt * 16 + l15;
      C[(size_t)(row0 + rr) * NC + col] = acc[nt][r] + bias[col];
    }
}

// ---------------- LayerNorm + head split (+ optional beta) -> bf16 [b][h][len][64] ----------------
__global__ __launch_bounds__(256) void ln_to_heads(
    const float* __restrict__ X, const float* __restrict__ w,
    const float* __restrict__ b, const float* __restrict__ scaling,
    unsigned short* __restrict__ out, int rows, int len) {
  int row = blockIdx.x * 8 + (threadIdx.x >> 5);
  int lane = threadIdx.x & 31;
  if (row >= rows) return;
  const float* x = X + (size_t)row * E_;
  float vals[16];
  float s = 0.f, ss = 0.f;
#pragma unroll
  for (int j = 0; j < 16; ++j) {
    float v = x[lane + 32 * j];
    vals[j] = v; s += v; ss += v * v;
  }
#pragma unroll
  for (int m = 1; m < 32; m <<= 1) {
    s += __shfl_xor(s, m, 32);
    ss += __shfl_xor(ss, m, 32);
  }
  float mean = s * (1.f / E_);
  float var = ss * (1.f / E_) - mean * mean;
  float rstd = rsqrtf(var + 1e-5f);
  int l = row / N_, n = row % N_;
#pragma unroll
  for (int j = 0; j < 16; ++j) {
    int c = lane + 32 * j;
    float y = (vals[j] - mean) * rstd * w[c] + b[c];
    int h = c >> 6, d = c & 63;
    if (scaling) y *= scaling[h];
    out[(((size_t)(n * H_ + h) * len + l) << 6) + d] = f2bf(y);
  }
}

// ---------------- Hopfield attention: 3 flash passes per 128-row q block ----------------
// k/v tiles double-buffered via TDM (tensor_load_to_lds + s_wait_tensorcnt).
__global__ __launch_bounds__(256) void hopfield_attn(
    const unsigned short* __restrict__ qh,   // [32][L][64] bf16 (pre-scaled by beta)
    const unsigned short* __restrict__ kh,   // [32][S][64]
    const unsigned short* __restrict__ vh,   // [32][S][64]
    const float* __restrict__ scaling,       // [8]
    float* __restrict__ AO) {                // [L][N][E] f32
  __shared__ unsigned short qs[128 * 64];     // 16KB current q block (bf16)
  __shared__ unsigned short ks[2][32 * 64];   // 8KB  double-buffered k tiles
  __shared__ unsigned short vs[2][32 * 64];   // 8KB  double-buffered v tiles
  __shared__ unsigned short ps[8][16 * 32];   // 8KB  per-wave softmax p

  int tid = threadIdx.x, wave = tid >> 5, lane = tid & 31;
  int half = lane >> 4, l15 = lane & 15;
  int kb8 = half * 8, kb16 = half * 16;
  int bh = blockIdx.y;
  int h = bh & (H_ - 1), n = bh >> 3;
  int l0 = blockIdx.x * 128;
  float beta = scaling[h];

  const unsigned short* qg = qh + ((size_t)bh * L_ + l0) * Dh_;
  const unsigned short* kg = kh + (size_t)bh * S_ * Dh_;
  const unsigned short* vg = vh + (size_t)bh * S_ * Dh_;

  for (int t = tid; t < 1024; t += 256)
    ((uint4*)qs)[t] = ((const uint4*)qg)[t];
  __syncthreads();

  int m0 = wave * 16;
  v8u qa[2];
  auto loadQA = [&]() {
#pragma unroll
    for (int kk = 0; kk < 2; ++kk)
#pragma unroll
      for (int j = 0; j < 8; ++j) {
        int e = (j < 4) ? (kb8 + 2 * j) : (16 + kb8 + 2 * (j - 4));
        qa[kk][j] = *(const unsigned int*)&qs[(m0 + l15) * 64 + kk * 32 + e];
      }
  };
  loadQA();

  v8f zf = {0.f, 0.f, 0.f, 0.f, 0.f, 0.f, 0.f, 0.f};
  const int NT = S_ / 32;   // 128 tiles

  for (int it = 0; it < 3; ++it) {
    float rm[8], rl[8];
    v8f acc[4];
    acc[0] = zf; acc[1] = zf; acc[2] = zf; acc[3] = zf;
#pragma unroll
    for (int r = 0; r < 8; ++r) { rm[r] = -1e30f; rl[r] = 0.f; }

    // prologue: DMA tile 0 (one contiguous 4KB block = 512 x 8B)
    if (wave == 0) {
      tdm_load(lds_off(&ks[0][0]), kg, 512u, 1u, 512u);
      if (it == 2) tdm_load(lds_off(&vs[0][0]), vg, 512u, 1u, 512u);
    }

    for (int st = 0; st < NT; ++st) {
      int buf = st & 1;
      __syncthreads();           // compute on buf^1 finished -> safe to overwrite
      if (wave == 0) {
        if (st + 1 < NT) {
          tdm_load(lds_off(&ks[buf ^ 1][0]), kg + (size_t)(st + 1) * 2048, 512u, 1u, 512u);
          if (it == 2) {
            tdm_load(lds_off(&vs[buf ^ 1][0]), vg + (size_t)(st + 1) * 2048, 512u, 1u, 512u);
            __builtin_amdgcn_s_wait_tensorcnt(2);   // tile st done, st+1 (k+v) in flight
          } else {
            __builtin_amdgcn_s_wait_tensorcnt(1);   // tile st done, st+1 in flight
          }
        } else {
          __builtin_amdgcn_s_wait_tensorcnt(0);
        }
      }
      __syncthreads();           // tile st visible to all waves

      const unsigned short* ksb = &ks[buf][0];
      const unsigned short* vsb = &vs[buf][0];

      // scores s = q @ k^T : B read transposed straight out of the LDS k tile
      v8f sc[2];
      sc[0] = zf; sc[1] = zf;
#pragma unroll
      for (int nt = 0; nt < 2; ++nt)
#pragma unroll
        for (int kk = 0; kk < 2; ++kk) {
          v8u bv;
#pragma unroll
          for (int j = 0; j < 8; ++j)
            bv[j] = *(const unsigned int*)&ksb[(nt * 16 + l15) * 64 + kk * 32 + ((j < 4) ? (kb8 + 2 * j) : (16 + kb8 + 2 * (j - 4)))];
          sc[nt] = wmma_bf16f32(qa[kk], bv, sc[nt]);
        }

      // online softmax (row stats replicated across each 16-lane half)
#pragma unroll
      for (int r = 0; r < 8; ++r) {
        float mx = fmaxf(sc[0][r], sc[1][r]);
#pragma unroll
        for (int m = 1; m < 16; m <<= 1) mx = fmaxf(mx, __shfl_xor(mx, m, 32));
        float nm = fmaxf(rm[r], mx);
        float scl = __expf(rm[r] - nm);
        rm[r] = nm;
        rl[r] *= scl;
#pragma unroll
        for (int dt = 0; dt < 4; ++dt) acc[dt][r] *= scl;
        float p0 = __expf(sc[0][r] - nm);
        float p1 = __expf(sc[1][r] - nm);
        sc[0][r] = p0; sc[1][r] = p1;
        float sum = p0 + p1;
#pragma unroll
        for (int m = 1; m < 16; m <<= 1) sum += __shfl_xor(sum, m, 32);
        rl[r] += sum;
      }

      // C-layout p -> bf16 in per-wave LDS, then re-read as A fragment
#pragma unroll
      for (int nt = 0; nt < 2; ++nt)
#pragma unroll
        for (int r = 0; r < 8; ++r)
          ps[wave][(r + half * 8) * 32 + nt * 16 + l15] = f2bf(sc[nt][r]);

      v8u pa;
#pragma unroll
      for (int j = 0; j < 8; ++j) {
        int e = (j < 4) ? (kb8 + 2 * j) : (16 + kb8 + 2 * (j - 4));
        pa[j] = *(const unsigned int*)&ps[wave][l15 * 32 + e];
      }

      // acc += p @ k   (iters 0,1)  or  p @ v  (iter 2)
      const unsigned short* src = (it == 2) ? vsb : ksb;
#pragma unroll
      for (int dt = 0; dt < 4; ++dt) {
        v8u bv;
#pragma unroll
        for (int j = 0; j < 8; ++j) {
          int e = (j < 4) ? (kb8 + 2 * j) : (16 + kb8 + 2 * (j - 4));
          unsigned int lo = src[e * 64 + dt * 16 + l15];
          unsigned int hi = src[(e + 1) * 64 + dt * 16 + l15];
          bv[j] = lo | (hi << 16);
        }
        acc[dt] = wmma_bf16f32(pa, bv, acc[dt]);
      }
    }  // S tiles

    if (it < 2) {
      // q <- softmax(s) @ k * beta  (normalize, requantize into our own LDS rows)
#pragma unroll
      for (int r = 0; r < 8; ++r) {
        float inv = beta / rl[r];
#pragma unroll
        for (int dt = 0; dt < 4; ++dt)
          qs[(m0 + r + half * 8) * 64 + dt * 16 + l15] = f2bf(acc[dt][r] * inv);
      }
      loadQA();
    } else {
      // final out -> [l][n][h*64+d] f32
#pragma unroll
      for (int r = 0; r < 8; ++r) {
        float inv = 1.f / rl[r];
        int l = l0 + m0 + r + half * 8;
#pragma unroll
        for (int dt = 0; dt < 4; ++dt) {
          int col = h * 64 + dt * 16 + l15;
          AO[((size_t)l * N_ + n) * E_ + col] = acc[dt][r] * inv;
        }
      }
    }
  }  // iterations
}

// ---------------- host orchestration ----------------
extern "C" void kernel_launch(void* const* d_in, const int* in_sizes, int n_in,
                              void* d_out, int out_size, void* d_ws, size_t ws_size,
                              hipStream_t stream) {
  const float* query = (const float*)d_in[0];
  const float* key   = (const float*)d_in[1];
  const float* value = (const float*)d_in[2];
  const float* ipw   = (const float*)d_in[3];   // (3E, E)
  const float* ipb   = (const float*)d_in[4];   // (3E,)
  const float* opw   = (const float*)d_in[5];   // (E, E)
  const float* opb   = (const float*)d_in[6];   // (E,)
  const float* lnw   = (const float*)d_in[7];
  const float* lnb   = (const float*)d_in[8];
  const float* scal  = (const float*)d_in[9];   // (H,)

  char* ws = (char*)d_ws;
  const size_t MB = 1u << 20;
  // bf16 input copies; same regions later reused as qh/kh/vh (consumed before overwrite)
  unsigned short* Xq  = (unsigned short*)(ws);                 //  2 MB
  unsigned short* Xk  = (unsigned short*)(ws + 2 * MB);        // 16 MB
  unsigned short* Xv  = (unsigned short*)(ws + 18 * MB);       // 16 MB
  unsigned short* Wip = (unsigned short*)(ws + 34 * MB);       // 1.5 MB (3E x E bf16)
  unsigned short* Wop = (unsigned short*)(ws + 35 * MB + 512 * 1024);  // .5 MB
  float* Pq = (float*)(ws + 36 * MB);                          //  4 MB, later AO
  float* Pk = (float*)(ws + 40 * MB);                          // 32 MB, head reused as AObf
  float* Pv = (float*)(ws + 72 * MB);                          // 32 MB

  const int nq = L_ * N_ * E_;        // 1,048,576
  const int ns = S_ * N_ * E_;        // 8,388,608
  dim3 blk(256);

  // bf16 conversions (weights stay row-major: LDS B tile [n][k] == W tile)
  cvt_f32_bf16<<<dim3((nq + 255) / 256), blk, 0, stream>>>(query, Xq, nq);
  cvt_f32_bf16<<<dim3((ns + 255) / 256), blk, 0, stream>>>(key,   Xk, ns);
  cvt_f32_bf16<<<dim3((ns + 255) / 256), blk, 0, stream>>>(value, Xv, ns);
  cvt_f32_bf16<<<dim3(3 * E_ * E_ / 256), blk, 0, stream>>>(ipw, Wip, 3 * E_ * E_);
  cvt_f32_bf16<<<dim3(E_ * E_ / 256),     blk, 0, stream>>>(opw, Wop, E_ * E_);

  // input projections (WMMA GEMMs, TDM-staged weight tiles)
  gemm_bf16<<<dim3(16, 8),  blk, 0, stream>>>(Xq, Wip,               ipb,          Pq, L_ * N_);
  gemm_bf16<<<dim3(128, 8), blk, 0, stream>>>(Xk, Wip + E_ * E_,     ipb + E_,     Pk, S_ * N_);
  gemm_bf16<<<dim3(128, 8), blk, 0, stream>>>(Xv, Wip + 2 * E_ * E_, ipb + 2 * E_, Pv, S_ * N_);

  // layernorm + head split (+ beta on q); overwrite the consumed bf16 input copies
  ln_to_heads<<<dim3(L_ * N_ / 8), blk, 0, stream>>>(Pq, lnw, lnb, scal,    Xq, L_ * N_, L_);
  ln_to_heads<<<dim3(S_ * N_ / 8), blk, 0, stream>>>(Pk, lnw, lnb, nullptr, Xk, S_ * N_, S_);
  ln_to_heads<<<dim3(S_ * N_ / 8), blk, 0, stream>>>(Pv, lnw, lnb, nullptr, Xv, S_ * N_, S_);

  // 3-pass Hopfield attention; AO reuses Pq (already consumed)
  float* AO = Pq;
  hopfield_attn<<<dim3(L_ / 128, N_ * H_), blk, 0, stream>>>(Xq, Xk, Xv, scal, AO);

  // output projection
  unsigned short* AObf = (unsigned short*)Pk;   // Pk consumed by LN already
  cvt_f32_bf16<<<dim3((nq + 255) / 256), blk, 0, stream>>>(AO, AObf, nq);
  gemm_bf16<<<dim3(16, 8), blk, 0, stream>>>(AObf, Wop, opb, (float*)d_out, L_ * N_);
}
